// SeparateWindowedAttention_33784212750722
// MI455X (gfx1250) — matrix operand order repaired and checked by reading, not compiled, not run on hardware
//
#include <hip/hip_runtime.h>
#include <hip/hip_bf16.h>

typedef __attribute__((ext_vector_type(16))) __bf16 v16bf;
typedef __attribute__((ext_vector_type(8)))  float  v8f;

constexpr int NTOK = 131072;   // B*H*W = 2*256*256
constexpr int NWIN = 2048;     // B*32*32 windows
constexpr int QKK  = 96;       // 64 feat + 16 coord + 16 pad

// ---------------------------------------------------------------------------
// CDNA5 async global->LDS copy (tracked with ASYNCcnt, no VGPR staging).
// vdst = LDS byte address, vaddr = per-lane byte offset, saddr = uniform base.
// ---------------------------------------------------------------------------
__device__ __forceinline__ void async_b128(unsigned int lds_addr,
                                           unsigned int byte_off,
                                           const void* base)
{
    asm volatile("global_load_async_to_lds_b128 %0, %1, %2"
                 :: "v"(lds_addr), "v"(byte_off), "s"(base)
                 : "memory");
}
__device__ __forceinline__ void wait_async0()
{
    asm volatile("s_wait_asynccnt 0x0" ::: "memory");
}
__device__ __forceinline__ unsigned int lds_off(const void* p)
{
    return (unsigned int)(unsigned long long)p;   // LDS aperture: low 32 bits
}

// ---------------------------------------------------------------------------
// fp32 -> bf16 conversion (vectorized)
// ---------------------------------------------------------------------------
__global__ void __launch_bounds__(256)
f32_to_bf16_k(const float* __restrict__ src, __bf16* __restrict__ dst, size_t n4)
{
    size_t i = (size_t)blockIdx.x * blockDim.x + threadIdx.x;
    if (i < n4) {
        float4 v = ((const float4*)src)[i];
        size_t b = i * 4;
        dst[b + 0] = (__bf16)v.x;
        dst[b + 1] = (__bf16)v.y;
        dst[b + 2] = (__bf16)v.z;
        dst[b + 3] = (__bf16)v.w;
    }
}

// ---------------------------------------------------------------------------
// relative position bias table: bias[p][q] = pos_emb[px-qx+7][py-qy+7]
// ---------------------------------------------------------------------------
__global__ void __launch_bounds__(64)
build_bias(const float* __restrict__ pos_emb, float* __restrict__ bias64)
{
    int p = blockIdx.x, q = threadIdx.x;
    int rx = (p >> 3) - (q >> 3) + 7;
    int ry = (p & 7)  - (q & 7)  + 7;
    bias64[p * 64 + q] = pos_emb[rx * 15 + ry];
}

// ---------------------------------------------------------------------------
// Tiled bf16 WMMA GEMM:  out[M,N] = A[M,K] @ W[K,N] + bias[N]
// block = 128 threads (4 waves), tile 64x64, K-step 32.
// Double-buffered LDS tiles filled with async global->LDS DMA.
// ---------------------------------------------------------------------------
template <bool OUT_F32>
__global__ void __launch_bounds__(128)
gemm_bf16(const __bf16* __restrict__ A, const __bf16* __restrict__ W,
          const float* __restrict__ bias, void* __restrict__ outp,
          int M, int N, int K)
{
    __shared__ __align__(16) __bf16 As[2][64][40];
    __shared__ __align__(16) __bf16 Ws[2][32][72];

    const int tid  = threadIdx.x;
    const int lane = tid & 31, wave = tid >> 5;
    const int half = lane >> 4, lm = lane & 15;
    const int tm = blockIdx.y * 64, tn = blockIdx.x * 64;

    const int arow = tid >> 1, acol = (tid & 1) * 16;
    const int wrow = tid >> 2, wcol = (tid & 3) * 16;

    v8f acc[4] = {};

    auto issue_tile = [&](int buf, int k0) {
        unsigned int la = lds_off(&As[buf][arow][acol]);
        unsigned int ga = (unsigned int)(((size_t)(tm + arow) * K + k0 + acol) * 2);
        async_b128(la,      ga,      A);
        async_b128(la + 16, ga + 16, A);
        unsigned int lw = lds_off(&Ws[buf][wrow][wcol]);
        unsigned int gw = (unsigned int)(((size_t)(k0 + wrow) * N + tn + wcol) * 2);
        async_b128(lw,      gw,      W);
        async_b128(lw + 16, gw + 16, W);
    };

    issue_tile(0, 0);

    int buf = 0;
    for (int k0 = 0; k0 < K; k0 += 32, buf ^= 1) {
        wait_async0();
        __syncthreads();
        // All waves finished computing on buf^1 before this barrier, so it is
        // safe to start the DMA for the next tile into buf^1 now.
        if (k0 + 32 < K) issue_tile(buf ^ 1, k0 + 32);

        // A fragment: 16x32, rows = wave strip, contiguous k = 16*half + i
        v16bf a;
#pragma unroll
        for (int i = 0; i < 16; ++i)
            a[i] = As[buf][wave * 16 + lm][half * 16 + i];
#pragma unroll
        for (int t = 0; t < 4; ++t) {
            v16bf b;
#pragma unroll
            for (int i = 0; i < 16; ++i)
                b[i] = Ws[buf][half * 16 + i][t * 16 + lm];
            acc[t] = __builtin_amdgcn_wmma_f32_16x16x32_bf16(
                false, a, false, b, (short)0, acc[t], false, false);
        }
    }

#pragma unroll
    for (int t = 0; t < 4; ++t)
#pragma unroll
        for (int r = 0; r < 8; ++r) {
            int row = tm + wave * 16 + half * 8 + r;
            int col = tn + t * 16 + lm;
            float v = acc[t][r] + bias[col];
            if (OUT_F32) ((float*)outp)[(size_t)row * N + col] = v;
            else         ((__bf16*)outp)[(size_t)row * N + col] = (__bf16)v;
        }
}

// ---------------------------------------------------------------------------
// RMSNorm + qkv scatter into windowed layout.
// concat channel c = 3d+k; c<1024 -> rmsnorm(f@Wf+bf)*gf, else v. d = e*8+head.
// q rows packed as [f_q * HD^-0.5 | alpha*c_q/sqrt(128) | zeros] (K = 96)
// ---------------------------------------------------------------------------
__global__ void __launch_bounds__(128)
scatter_qkv(const __bf16* __restrict__ rfqk, const __bf16* __restrict__ rv,
            const __bf16* __restrict__ rcqk,
            const float* __restrict__ gf, const float* __restrict__ gc,
            const float* __restrict__ alpha,
            __bf16* __restrict__ qext, __bf16* __restrict__ kext,
            __bf16* __restrict__ vt)
{
    __shared__ float fq[1024];
    __shared__ float vv[512];
    __shared__ float cq[256];
    __shared__ float red[128], redc[128];
    __shared__ float sfv, scv;

    const int tid = threadIdx.x;
    const size_t t = blockIdx.x;

    float pf = 0.f, pc = 0.f;
#pragma unroll
    for (int j = 0; j < 8; ++j) {
        int c = tid * 8 + j;
        float x = (float)rfqk[t * 1024 + c];
        pf += x * x;
        fq[c] = x * gf[c];
    }
#pragma unroll
    for (int j = 0; j < 4; ++j) {
        int c = tid * 4 + j;
        vv[c] = (float)rv[t * 512 + c];
    }
#pragma unroll
    for (int j = 0; j < 2; ++j) {
        int c = tid * 2 + j;
        float x = (float)rcqk[t * 256 + c];
        pc += x * x;
        cq[c] = x * gc[c];
    }
    red[tid] = pf; redc[tid] = pc;
    __syncthreads();
    for (int s = 64; s > 0; s >>= 1) {
        if (tid < s) { red[tid] += red[tid + s]; redc[tid] += redc[tid + s]; }
        __syncthreads();
    }
    if (tid == 0) {
        sfv = rsqrtf(red[0]  / 1024.f + 1e-6f);
        scv = rsqrtf(redc[0] / 256.f  + 1e-6f);
    }
    __syncthreads();

    const int bimg = (int)(t >> 16);
    const int h = ((int)t >> 8) & 255, w = (int)t & 255;
    const int win = ((bimg * 32) + (h >> 3)) * 32 + (w >> 3);
    const int p = (h & 7) * 8 + (w & 7);
    const float s_f = sfv, s_c = scv;
    const float invsq128 = 0.08838834764831845f;  // 1/sqrt(128)

#pragma unroll
    for (int loop = 0; loop < 4; ++loop) {
        int d = tid + loop * 128;            // d = e*8 + head
        int head = d & 7, e = d >> 3;
        int c0 = 3 * d;
        float qv = (c0     < 1024 ? fq[c0]     * s_f : vv[c0 - 1024]) * 0.125f;
        float kv = (c0 + 1 < 1024 ? fq[c0 + 1] * s_f : vv[c0 + 1 - 1024]);
        float vx = (c0 + 2 < 1024 ? fq[c0 + 2] * s_f : vv[c0 + 2 - 1024]);
        size_t base = (size_t)(win * 8 + head) * 64 + p;
        qext[base * QKK + e] = (__bf16)qv;
        kext[base * QKK + e] = (__bf16)kv;
        vt[base * 64 + e]    = (__bf16)vx;
    }
    {   // coords (d = e*8+head, e<16, channels 2d / 2d+1) + zero padding
        int d = tid;
        int head = d & 7, e = d >> 3;
        size_t base = (size_t)(win * 8 + head) * 64 + p;
        float qv = cq[2 * d]     * s_c * alpha[head] * invsq128;
        float kv = cq[2 * d + 1] * s_c;
        qext[base * QKK + 64 + e] = (__bf16)qv;
        kext[base * QKK + 64 + e] = (__bf16)kv;
        int head2 = tid >> 4, e2 = 80 + (tid & 15);
        size_t base2 = (size_t)(win * 8 + head2) * 64 + p;
        qext[base2 * QKK + e2] = (__bf16)0.f;
        kext[base2 * QKK + e2] = (__bf16)0.f;
    }
}

// ---------------------------------------------------------------------------
// Windowed attention: one block per (window, head). 64 tokens, fused
// feature+coord dots via K=96, softmax, att@V, scatter to (H e) layout.
// Q/K/V tiles pulled into LDS with async DMA.
// ---------------------------------------------------------------------------
__global__ void __launch_bounds__(128)
win_attn(const __bf16* __restrict__ qext, const __bf16* __restrict__ kext,
         const __bf16* __restrict__ vt, const float* __restrict__ bias64,
         __bf16* __restrict__ ybf)
{
    __shared__ __align__(16) __bf16 Qs[64][QKK];
    __shared__ __align__(16) __bf16 Ks[64][QKK];
    __shared__ __align__(16) __bf16 Vs[64][64];
    __shared__ float  Ds[64][64];
    __shared__ __bf16 Am[64][64];

    const int tid  = threadIdx.x;
    const int lane = tid & 31, wave = tid >> 5;
    const int half = lane >> 4, lm = lane & 15;
    const int bh = blockIdx.x;           // win*8 + head
    const int head = bh & 7, win = bh >> 3;

    {   // async global->LDS fills (contiguous layouts)
        const __bf16* qg = qext + (size_t)bh * (64 * QKK);
        const __bf16* kg = kext + (size_t)bh * (64 * QKK);
        const __bf16* vg = vt   + (size_t)bh * (64 * 64);
        unsigned int lq = lds_off(&Qs[0][0]);
        unsigned int lk = lds_off(&Ks[0][0]);
        unsigned int lv = lds_off(&Vs[0][0]);
#pragma unroll
        for (int j = 0; j < 6; ++j) {
            unsigned int o = (unsigned int)(tid + 128 * j) * 16;
            async_b128(lq + o, o, qg);
        }
#pragma unroll
        for (int j = 0; j < 6; ++j) {
            unsigned int o = (unsigned int)(tid + 128 * j) * 16;
            async_b128(lk + o, o, kg);
        }
#pragma unroll
        for (int j = 0; j < 4; ++j) {
            unsigned int o = (unsigned int)(tid + 128 * j) * 16;
            async_b128(lv + o, o, vg);
        }
    }
    wait_async0();
    __syncthreads();

    // dots(64x64) = Q(64x96) @ K^T   (3 k-steps of 32)
    v8f acc[4] = {};
#pragma unroll
    for (int ks = 0; ks < 3; ++ks) {
        int k0 = ks * 32;
        v16bf a;
#pragma unroll
        for (int i = 0; i < 16; ++i)
            a[i] = Qs[wave * 16 + lm][k0 + half * 16 + i];
#pragma unroll
        for (int t = 0; t < 4; ++t) {
            v16bf b;   // B[e][q] = Ks[q][e]
#pragma unroll
            for (int i = 0; i < 16; ++i)
                b[i] = Ks[t * 16 + lm][k0 + half * 16 + i];
            acc[t] = __builtin_amdgcn_wmma_f32_16x16x32_bf16(
                false, a, false, b, (short)0, acc[t], false, false);
        }
    }
#pragma unroll
    for (int t = 0; t < 4; ++t)
#pragma unroll
        for (int r = 0; r < 8; ++r) {
            int row = wave * 16 + half * 8 + r, col = t * 16 + lm;
            Ds[row][col] = acc[t][r] + bias64[row * 64 + col];
        }
    __syncthreads();

    if (tid < 64) {   // row softmax
        float m = -1e30f;
        for (int q = 0; q < 64; ++q) m = fmaxf(m, Ds[tid][q]);
        float s = 0.f;
        for (int q = 0; q < 64; ++q) { float e = __expf(Ds[tid][q] - m); Ds[tid][q] = e; s += e; }
        float inv = 1.f / s;
        for (int q = 0; q < 64; ++q) Am[tid][q] = (__bf16)(Ds[tid][q] * inv);
    }
    __syncthreads();

    // y(64x64) = att @ V   (2 k-steps of 32)
    v8f acc2[4] = {};
#pragma unroll
    for (int ks = 0; ks < 2; ++ks) {
        int k0 = ks * 32;
        v16bf a;
#pragma unroll
        for (int i = 0; i < 16; ++i)
            a[i] = Am[wave * 16 + lm][k0 + half * 16 + i];
#pragma unroll
        for (int t = 0; t < 4; ++t) {
            v16bf b;   // B[q][e] = Vs[q][e]
#pragma unroll
            for (int i = 0; i < 16; ++i)
                b[i] = Vs[k0 + half * 16 + i][t * 16 + lm];
            acc2[t] = __builtin_amdgcn_wmma_f32_16x16x32_bf16(
                false, a, false, b, (short)0, acc2[t], false, false);
        }
    }

    const int bimg = win >> 10, wi = (win >> 5) & 31, wj = win & 31;
#pragma unroll
    for (int t = 0; t < 4; ++t)
#pragma unroll
        for (int r = 0; r < 8; ++r) {
            int p = wave * 16 + half * 8 + r, e = t * 16 + lm;
            int h = wi * 8 + (p >> 3), w = wj * 8 + (p & 7);
            size_t tok = ((size_t)bimg * 256 + h) * 256 + w;
            ybf[tok * 512 + head * 64 + e] = (__bf16)acc2[t][r];   // '(H e)' layout
        }
}

// ---------------------------------------------------------------------------
extern "C" void kernel_launch(void* const* d_in, const int* in_sizes, int n_in,
                              void* d_out, int out_size, void* d_ws, size_t ws_size,
                              hipStream_t stream)
{
    (void)in_sizes; (void)n_in; (void)out_size; (void)ws_size;
    const float* features = (const float*)d_in[0];
    const float* coords   = (const float*)d_in[1];
    const float* Wf = (const float*)d_in[2];
    const float* bf = (const float*)d_in[3];
    const float* gf = (const float*)d_in[4];
    const float* Wc = (const float*)d_in[5];
    const float* bc = (const float*)d_in[6];
    const float* gc = (const float*)d_in[7];
    const float* Wv = (const float*)d_in[8];
    const float* bv = (const float*)d_in[9];
    const float* Wo = (const float*)d_in[10];
    const float* bo = (const float*)d_in[11];
    const float* alpha = (const float*)d_in[12];
    const float* pos_emb = (const float*)d_in[13];

    char* base = (char*)d_ws;
    size_t off = 0;
    auto alloc = [&](size_t bytes) -> void* {
        void* r = base + off;
        off = (off + bytes + 255) & ~(size_t)255;
        return r;
    };
    __bf16* feat_bf = (__bf16*)alloc((size_t)NTOK * 512 * 2);
    __bf16* coor_bf = (__bf16*)alloc((size_t)NTOK * 128 * 2);
    __bf16* Wf_bf   = (__bf16*)alloc((size_t)512 * 1024 * 2);
    __bf16* Wc_bf   = (__bf16*)alloc((size_t)128 * 256 * 2);
    __bf16* Wv_bf   = (__bf16*)alloc((size_t)512 * 512 * 2);
    __bf16* Wo_bf   = (__bf16*)alloc((size_t)512 * 512 * 2);
    __bf16* rfqk    = (__bf16*)alloc((size_t)NTOK * 1024 * 2);
    __bf16* rv      = (__bf16*)alloc((size_t)NTOK * 512 * 2);
    __bf16* rcqk    = (__bf16*)alloc((size_t)NTOK * 256 * 2);
    __bf16* qext    = (__bf16*)alloc((size_t)NWIN * 8 * 64 * QKK * 2);
    __bf16* kext    = (__bf16*)alloc((size_t)NWIN * 8 * 64 * QKK * 2);
    __bf16* vt      = (__bf16*)alloc((size_t)NWIN * 8 * 64 * 64 * 2);
    __bf16* ybf     = (__bf16*)alloc((size_t)NTOK * 512 * 2);
    float*  bias64  = (float*)alloc(64 * 64 * 4);

    auto conv = [&](const float* s, __bf16* d, size_t n) {
        size_t n4 = n / 4;
        f32_to_bf16_k<<<dim3((unsigned)((n4 + 255) / 256)), dim3(256), 0, stream>>>(s, d, n4);
    };
    conv(features, feat_bf, (size_t)NTOK * 512);
    conv(coords,   coor_bf, (size_t)NTOK * 128);
    conv(Wf, Wf_bf, (size_t)512 * 1024);
    conv(Wc, Wc_bf, (size_t)128 * 256);
    conv(Wv, Wv_bf, (size_t)512 * 512);
    conv(Wo, Wo_bf, (size_t)512 * 512);

    build_bias<<<dim3(64), dim3(64), 0, stream>>>(pos_emb, bias64);

    // raw projections (pre-norm), bf16 out
    gemm_bf16<false><<<dim3(1024 / 64, NTOK / 64), dim3(128), 0, stream>>>(
        feat_bf, Wf_bf, bf, rfqk, NTOK, 1024, 512);
    gemm_bf16<false><<<dim3(512 / 64, NTOK / 64), dim3(128), 0, stream>>>(
        feat_bf, Wv_bf, bv, rv, NTOK, 512, 512);
    gemm_bf16<false><<<dim3(256 / 64, NTOK / 64), dim3(128), 0, stream>>>(
        coor_bf, Wc_bf, bc, rcqk, NTOK, 256, 128);

    scatter_qkv<<<dim3(NTOK), dim3(128), 0, stream>>>(
        rfqk, rv, rcqk, gf, gc, alpha, qext, kext, vt);

    win_attn<<<dim3(NWIN * 8), dim3(128), 0, stream>>>(
        qext, kext, vt, bias64, ybf);

    // final projection, fp32 out
    gemm_bf16<true><<<dim3(512 / 64, NTOK / 64), dim3(128), 0, stream>>>(
        ybf, Wo_bf, bo, d_out, NTOK, 512, 512);
}
